// RSSM_412316860803
// MI455X (gfx1250) — compile-verified, matrix operand order checked by reading, not run on hardware
//
#include <hip/hip_runtime.h>

typedef __bf16 bf16;
typedef __attribute__((ext_vector_type(16))) __bf16 v16bf;
typedef __attribute__((ext_vector_type(8)))  float  v8f;

#define T_STEPS 64
#define B_DIM 256
#define S_DIM 256
#define D_DIM 1024
#define H_DIM 1024
#define E_DIM 1536
#define A_DIM 6
#define KSA   320          // S+A padded to multiple of 64
#define OUT_STRIDE 2560    // 6*S + D

// Async copy of one 16B chunk: global memory -> LDS, tracked by ASYNCcnt.
__device__ __forceinline__ void async_copy16(unsigned int lds_addr, const void* gaddr)
{
    asm volatile("global_load_async_to_lds_b128 %0, %1, off"
                 :: "v"(lds_addr), "v"(gaddr) : "memory");
}
__device__ __forceinline__ void wait_async0()
{
    asm volatile("s_wait_asynccnt 0" ::: "memory");
}

// ---------------------------------------------------------------------------
// Tiled WMMA GEMM: C[M=256, N] = act( A[M,K] @ Bt[N,K]^T + bias )
// BM=128, BN=64, BK=64. 256 threads = 8 waves; wave grid 4(M)x2(N),
// each wave computes a 32x32 patch = 2x2 v_wmma_f32_16x16x32_bf16 tiles.
// Double-buffered LDS fed by global_load_async_to_lds_b128.
// ---------------------------------------------------------------------------
__global__ __launch_bounds__(256)
void gemm_wmma_bf16(const bf16* __restrict__ A, int lda,
                    const bf16* __restrict__ Bt, int ldb,
                    const float* __restrict__ bias,
                    float* __restrict__ Cf, int ldcf,
                    bf16* __restrict__ Cb, int ldcb,
                    int K, int act)
{
    __shared__ __align__(16) bf16 As[2][128 * 72];   // 64 K + 8 pad per row
    __shared__ __align__(16) bf16 Bs[2][64 * 72];

    const int tid  = threadIdx.x;
    const int lane = tid & 31;
    const int wid  = tid >> 5;
    const int wm   = wid >> 1;         // 0..3 -> 32-row strip
    const int wn   = wid & 1;          // 0..1 -> 32-col strip
    const int mBlock = blockIdx.y * 128;
    const int nBlock = blockIdx.x * 64;

    const int lr = lane & 15;          // row/col within 16-tile
    const int hi = lane >> 4;          // K-half select per ISA layout

    // Per-thread staging chunks: A tile 128x64 = 1024 16B-chunks (4/thread),
    // B tile 64x64 = 512 chunks (2/thread).
    unsigned int aoff[4], boff[2];
    const bf16 *ga[4], *gb[2];
#pragma unroll
    for (int i = 0; i < 4; ++i) {
        int c = tid + i * 256, row = c >> 3, kc = c & 7;
        aoff[i] = (unsigned int)((row * 72 + kc * 8) * 2);
        ga[i]   = A + (size_t)(mBlock + row) * lda + kc * 8;
    }
#pragma unroll
    for (int i = 0; i < 2; ++i) {
        int c = tid + i * 256, row = c >> 3, kc = c & 7;
        boff[i] = (unsigned int)((row * 72 + kc * 8) * 2);
        gb[i]   = Bt + (size_t)(nBlock + row) * ldb + kc * 8;
    }
    const unsigned int aBase[2] = { (unsigned int)(uintptr_t)&As[0][0],
                                    (unsigned int)(uintptr_t)&As[1][0] };
    const unsigned int bBase[2] = { (unsigned int)(uintptr_t)&Bs[0][0],
                                    (unsigned int)(uintptr_t)&Bs[1][0] };

    auto issue_stage = [&](int kb, int buf) {
#pragma unroll
        for (int i = 0; i < 4; ++i)
            async_copy16(aBase[buf] + aoff[i], ga[i] + kb * 64);
#pragma unroll
        for (int i = 0; i < 2; ++i)
            async_copy16(bBase[buf] + boff[i], gb[i] + kb * 64);
    };

    v8f acc[2][2] = {};
    const int nStages = K >> 6;

    issue_stage(0, 0);
    wait_async0();
    __syncthreads();

    for (int kb = 0; kb < nStages; ++kb) {
        const int buf = kb & 1;
        if (kb + 1 < nStages) issue_stage(kb + 1, buf ^ 1);   // prefetch next stage

        const bf16* asb = &As[buf][0];
        const bf16* bsb = &Bs[buf][0];
#pragma unroll
        for (int ks = 0; ks < 64; ks += 32) {
            v16bf af[2], bfr[2];
#pragma unroll
            for (int i = 0; i < 2; ++i) {
                // A fragment: lanes 0-15 hold K 0..7 & 16..23, lanes 16-31 K 8..15 & 24..31
                const bf16* pa = asb + (wm * 32 + i * 16 + lr) * 72 + ks + hi * 8;
                union { uint4 u[2]; v16bf v; } ca;
                ca.u[0] = *(const uint4*)(pa);
                ca.u[1] = *(const uint4*)(pa + 16);
                af[i] = ca.v;
                // B fragment: identical gather (Bt is column-major of original W)
                const bf16* pb = bsb + (wn * 32 + i * 16 + lr) * 72 + ks + hi * 8;
                union { uint4 u[2]; v16bf v; } cb;
                cb.u[0] = *(const uint4*)(pb);
                cb.u[1] = *(const uint4*)(pb + 16);
                bfr[i] = cb.v;
            }
#pragma unroll
            for (int i = 0; i < 2; ++i)
#pragma unroll
                for (int j = 0; j < 2; ++j)
                    acc[i][j] = __builtin_amdgcn_wmma_f32_16x16x32_bf16(
                        false, af[i], false, bfr[j], (short)0, acc[i][j], false, false);
        }
        wait_async0();        // next stage's async LDS writes (this wave) done
        __syncthreads();      // all waves done writing next buf + done reading this buf
    }

    // Epilogue: C/D layout — lane (0-15): N=lane, M=vgpr; lane (16-31): N=lane-16, M=vgpr+8
#pragma unroll
    for (int i = 0; i < 2; ++i) {
#pragma unroll
        for (int j = 0; j < 2; ++j) {
            int n = nBlock + wn * 32 + j * 16 + lr;
            float bv = bias ? bias[n] : 0.f;
#pragma unroll
            for (int v = 0; v < 8; ++v) {
                int m = mBlock + wm * 32 + i * 16 + v + hi * 8;
                float x = acc[i][j][v] + bv;
                if (act) x = x > 0.f ? x : (__expf(x) - 1.f);   // ELU
                if (Cf) Cf[(size_t)m * ldcf + n] = x;
                if (Cb) Cb[(size_t)m * ldcb + n] = (bf16)x;
            }
        }
    }
}

// ---------------------------------------------------------------------------
// Weight convert + transpose (fp32 KxN -> bf16 NxKpad, zero-padded K)
// ---------------------------------------------------------------------------
__global__ void wt_conv(const float* __restrict__ W, bf16* __restrict__ Wt,
                        int K, int N, int Kpad)
{
    size_t i = (size_t)blockIdx.x * 256 + threadIdx.x;
    if (i >= (size_t)N * Kpad) return;
    int n = (int)(i / Kpad), k = (int)(i % Kpad);
    float v = (k < K) ? W[(size_t)k * N + n] : 0.f;
    Wt[i] = (bf16)v;
}

__global__ void init_state(const float* __restrict__ is, const float* __restrict__ id,
                           float* __restrict__ stoch, float* __restrict__ deter)
{
    int i = blockIdx.x * 256 + threadIdx.x;
    if (i < B_DIM * S_DIM) stoch[i] = is[i];
    int j = i - B_DIM * S_DIM;
    if (j >= 0 && j < B_DIM * D_DIM) deter[j] = id[j];
}

// Per-step input prep: sa = [stoch*nt | act*nt | 0pad] -> bf16,
// h_bf16 = bf16(deter*nt), qx[:,1024:] = bf16(obs).
__global__ void prep_step(const float* __restrict__ stoch, const float* __restrict__ act_t,
                          const float* __restrict__ nt_t, const float* __restrict__ obs_t,
                          const float* __restrict__ deter,
                          bf16* __restrict__ sa, bf16* __restrict__ hbf, bf16* __restrict__ qx)
{
    int i = blockIdx.x * 256 + threadIdx.x;     // B * 2880
    int b = i / 2880, j = i % 2880;
    float nt = nt_t[b];
    if (j < KSA) {
        float v = 0.f;
        if (j < S_DIM)                 v = stoch[b * S_DIM + j] * nt;
        else if (j < S_DIM + A_DIM)    v = act_t[b * A_DIM + (j - S_DIM)] * nt;
        sa[b * KSA + j] = (bf16)v;
    } else if (j < KSA + D_DIM) {
        int d = j - KSA;
        hbf[b * D_DIM + d] = (bf16)(deter[b * D_DIM + d] * nt);
    } else {
        int e = j - (KSA + D_DIM);
        qx[(size_t)b * 2560 + 1024 + e] = (bf16)obs_t[b * E_DIM + e];
    }
}

// GRU gate math (gate order r,z,n). Writes new deter (f32 state + bf16 into qx[:, :1024])
// and the deter slice of the output.
__global__ void gru_gates(const float* __restrict__ gi, const float* __restrict__ gh,
                          const float* __restrict__ nt_t, float* __restrict__ deter,
                          bf16* __restrict__ qx, float* __restrict__ out_t)
{
    int i = blockIdx.x * 256 + threadIdx.x;     // B * D
    int b = i / D_DIM, d = i % D_DIM;
    size_t gb = (size_t)b * 3 * D_DIM;
    float ir = gi[gb + d], iz = gi[gb + D_DIM + d], in_ = gi[gb + 2 * D_DIM + d];
    float hr = gh[gb + d], hz = gh[gb + D_DIM + d], hn  = gh[gb + 2 * D_DIM + d];
    float r = 1.f / (1.f + __expf(-(ir + hr)));
    float z = 1.f / (1.f + __expf(-(iz + hz)));
    float n = tanhf(in_ + r * hn);
    float hprev = deter[i] * nt_t[b];
    float dn = (1.f - z) * n + z * hprev;
    deter[i] = dn;
    qx[(size_t)b * 2560 + d] = (bf16)dn;
    out_t[(size_t)b * OUT_STRIDE + 1536 + d] = dn;
}

// Distribution head: mean | softplus(raw)+MIN_STD | mean + std*noise.
__global__ void dist_out(const float* __restrict__ po, const float* __restrict__ nz_t,
                         float* __restrict__ out_t, int base, float* __restrict__ stoch)
{
    int i = blockIdx.x * 256 + threadIdx.x;     // B * S
    int b = i / S_DIM, s = i % S_DIM;
    float mean = po[(size_t)b * 2 * S_DIM + s];
    float raw  = po[(size_t)b * 2 * S_DIM + S_DIM + s];
    float sp   = raw > 20.f ? raw : log1pf(__expf(raw));
    float std_ = sp + 0.1f;
    float st   = mean + std_ * nz_t[i];
    float* o = out_t + (size_t)b * OUT_STRIDE + base;
    o[s] = mean; o[S_DIM + s] = std_; o[2 * S_DIM + s] = st;
    if (stoch) stoch[i] = st;
}

// ---------------------------------------------------------------------------
static inline size_t alignup(size_t x) { return (x + 255) & ~(size_t)255; }

extern "C" void kernel_launch(void* const* d_in, const int* in_sizes, int n_in,
                              void* d_out, int out_size, void* d_ws, size_t ws_size,
                              hipStream_t stream)
{
    const float* obs   = (const float*)d_in[0];
    const float* acts  = (const float*)d_in[1];
    const float* nts   = (const float*)d_in[2];
    const float* ist   = (const float*)d_in[3];
    const float* idt   = (const float*)d_in[4];
    const float* nzp   = (const float*)d_in[5];
    const float* nzq   = (const float*)d_in[6];
    const float* W_e   = (const float*)d_in[7];
    const float* b_e   = (const float*)d_in[8];
    const float* W_ih  = (const float*)d_in[9];
    const float* W_hh  = (const float*)d_in[10];
    const float* b_ih  = (const float*)d_in[11];
    const float* b_hh  = (const float*)d_in[12];
    const float* W_p1  = (const float*)d_in[13];
    const float* b_p1  = (const float*)d_in[14];
    const float* W_p2  = (const float*)d_in[15];
    const float* b_p2  = (const float*)d_in[16];
    const float* W_q1  = (const float*)d_in[17];
    const float* b_q1  = (const float*)d_in[18];
    const float* W_q2  = (const float*)d_in[19];
    const float* b_q2  = (const float*)d_in[20];
    float* out = (float*)d_out;

    char* w = (char*)d_ws;
    size_t off = 0;
    auto carve = [&](size_t bytes) { void* p = w + off; off = alignup(off + bytes); return p; };

    bf16* WeT  = (bf16*)carve((size_t)1024 * 320 * 2);
    bf16* WihT = (bf16*)carve((size_t)3072 * 1024 * 2);
    bf16* WhhT = (bf16*)carve((size_t)3072 * 1024 * 2);
    bf16* Wp1T = (bf16*)carve((size_t)1024 * 1024 * 2);
    bf16* Wp2T = (bf16*)carve((size_t)512  * 1024 * 2);
    bf16* Wq1T = (bf16*)carve((size_t)1024 * 2560 * 2);
    bf16* Wq2T = (bf16*)carve((size_t)512  * 1024 * 2);
    bf16* sa   = (bf16*)carve((size_t)B_DIM * KSA  * 2);
    bf16* hbf  = (bf16*)carve((size_t)B_DIM * D_DIM * 2);
    bf16* qx   = (bf16*)carve((size_t)B_DIM * 2560 * 2);
    bf16* emb  = (bf16*)carve((size_t)B_DIM * D_DIM * 2);
    bf16* ph   = (bf16*)carve((size_t)B_DIM * H_DIM * 2);
    bf16* qh   = (bf16*)carve((size_t)B_DIM * H_DIM * 2);
    float* gi   = (float*)carve((size_t)B_DIM * 3 * D_DIM * 4);
    float* gh   = (float*)carve((size_t)B_DIM * 3 * D_DIM * 4);
    float* pout = (float*)carve((size_t)B_DIM * 2 * S_DIM * 4);
    float* qout = (float*)carve((size_t)B_DIM * 2 * S_DIM * 4);
    float* deter= (float*)carve((size_t)B_DIM * D_DIM * 4);
    float* stoch= (float*)carve((size_t)B_DIM * S_DIM * 4);
    (void)ws_size; (void)n_in; (void)in_sizes; (void)out_size;

    // One-time (per launch) weight convert+transpose and state init.
    wt_conv<<<(1024 * 320)  / 256, 256, 0, stream>>>(W_e,  WeT,  262,  1024, 320);
    wt_conv<<<(3072 * 1024) / 256, 256, 0, stream>>>(W_ih, WihT, 1024, 3072, 1024);
    wt_conv<<<(3072 * 1024) / 256, 256, 0, stream>>>(W_hh, WhhT, 1024, 3072, 1024);
    wt_conv<<<(1024 * 1024) / 256, 256, 0, stream>>>(W_p1, Wp1T, 1024, 1024, 1024);
    wt_conv<<<(512  * 1024) / 256, 256, 0, stream>>>(W_p2, Wp2T, 1024, 512,  1024);
    wt_conv<<<(1024 * 2560) / 256, 256, 0, stream>>>(W_q1, Wq1T, 2560, 1024, 2560);
    wt_conv<<<(512  * 1024) / 256, 256, 0, stream>>>(W_q2, Wq2T, 1024, 512,  1024);
    init_state<<<(B_DIM * (S_DIM + D_DIM)) / 256, 256, 0, stream>>>(ist, idt, stoch, deter);

    for (int t = 0; t < T_STEPS; ++t) {
        const float* nt_t  = nts + (size_t)t * B_DIM;
        const float* act_t = acts + (size_t)t * B_DIM * A_DIM;
        const float* obs_t = obs + (size_t)t * B_DIM * E_DIM;
        const float* nzp_t = nzp + (size_t)t * B_DIM * S_DIM;
        const float* nzq_t = nzq + (size_t)t * B_DIM * S_DIM;
        float* out_t = out + (size_t)t * B_DIM * OUT_STRIDE;

        prep_step<<<B_DIM * 2880 / 256, 256, 0, stream>>>(stoch, act_t, nt_t, obs_t, deter,
                                                          sa, hbf, qx);
        // embed = elu(sa @ W_e + b_e)
        gemm_wmma_bf16<<<dim3(1024 / 64, 2), 256, 0, stream>>>(
            sa, KSA, WeT, 320, b_e, nullptr, 0, emb, 1024, 320, 1);
        // gi = embed @ W_ih + b_ih ; gh = h @ W_hh + b_hh
        gemm_wmma_bf16<<<dim3(3072 / 64, 2), 256, 0, stream>>>(
            emb, 1024, WihT, 1024, b_ih, gi, 3072, nullptr, 0, 1024, 0);
        gemm_wmma_bf16<<<dim3(3072 / 64, 2), 256, 0, stream>>>(
            hbf, 1024, WhhT, 1024, b_hh, gh, 3072, nullptr, 0, 1024, 0);
        gru_gates<<<B_DIM * D_DIM / 256, 256, 0, stream>>>(gi, gh, nt_t, deter, qx, out_t);
        // prior: ph = elu(deter @ W_p1 + b_p1); pout = ph @ W_p2 + b_p2
        gemm_wmma_bf16<<<dim3(1024 / 64, 2), 256, 0, stream>>>(
            qx, 2560, Wp1T, 1024, b_p1, nullptr, 0, ph, 1024, 1024, 1);
        gemm_wmma_bf16<<<dim3(512 / 64, 2), 256, 0, stream>>>(
            ph, 1024, Wp2T, 1024, b_p2, pout, 512, nullptr, 0, 1024, 0);
        dist_out<<<B_DIM * S_DIM / 256, 256, 0, stream>>>(pout, nzp_t, out_t, 0, nullptr);
        // posterior: qh = elu([deter|obs] @ W_q1 + b_q1); qout = qh @ W_q2 + b_q2
        gemm_wmma_bf16<<<dim3(1024 / 64, 2), 256, 0, stream>>>(
            qx, 2560, Wq1T, 2560, b_q1, nullptr, 0, qh, 1024, 2560, 1);
        gemm_wmma_bf16<<<dim3(512 / 64, 2), 256, 0, stream>>>(
            qh, 1024, Wq2T, 1024, b_q2, qout, 512, nullptr, 0, 1024, 0);
        dist_out<<<B_DIM * S_DIM / 256, 256, 0, stream>>>(qout, nzq_t, out_t, 3 * S_DIM, stoch);
    }
}
